// GAT_plm_76390288327214
// MI455X (gfx1250) — matrix operand order repaired and checked by reading, not compiled, not run on hardware
//
#include <hip/hip_runtime.h>
#include <hip/hip_bf16.h>
#include <math.h>

// ---------------------------------------------------------------------------
// Types for CDNA5 WMMA (wave32, 16x16x32 bf16 -> f32)
// ---------------------------------------------------------------------------
typedef __bf16 bf16;
typedef bf16     v16bf __attribute__((ext_vector_type(16)));
typedef float    v8f   __attribute__((ext_vector_type(8)));
typedef unsigned v8u   __attribute__((ext_vector_type(8)));

#define D_IN_C  1024
#define D_OUT_C 128
#define NGRAPH  128

// precise RNE (used once in weight reorder)
__device__ inline bf16 f2bf(float f) {
  unsigned u = __float_as_uint(f);
  unsigned r = u + 0x7FFFu + ((u >> 16) & 1u);
  unsigned short hs = (unsigned short)(r >> 16);
  bf16 out;
  __builtin_memcpy(&out, &hs, sizeof(hs));
  return out;
}

// fast pack: two f32 -> packed bf16x2 via one v_perm_b32 (round-half-up)
__device__ inline unsigned pack_bf2(float lo, float hi) {
  unsigned a = __float_as_uint(lo) + 0x8000u;
  unsigned b = __float_as_uint(hi) + 0x8000u;
  return __builtin_amdgcn_perm(b, a, 0x07060302u);  // {b[31:16], a[31:16]}
}

__device__ inline void atomicAddF(float* p, float v) {
  __hip_atomic_fetch_add(p, v, __ATOMIC_RELAXED, __HIP_MEMORY_SCOPE_AGENT);
}

__device__ inline void atomicMaxF(float* addr, float val) {
  unsigned* ua = (unsigned*)addr;
  unsigned old = *ua;
  while (true) {
    float cur = __uint_as_float(old);
    if (cur >= val) break;
    unsigned assumed = old;
    old = atomicCAS(ua, assumed, __float_as_uint(val));
    if (old == assumed) break;
  }
}

// ---------------------------------------------------------------------------
// Generic fill
// ---------------------------------------------------------------------------
__global__ void fill_f(float* __restrict__ p, float v, int n) {
  int i = blockIdx.x * 256 + threadIdx.x;
  if (i < n) p[i] = v;
}

// ---------------------------------------------------------------------------
// Reorder W (D_IN x D_OUT fp32, row-major) into bf16 B-fragment layout:
// wb[((kt*8 + nt)*32 + lane)*8 + v]  (uint32 = 2 packed bf16)
// B 32x16 layout: lanes 0-15 -> K {0..7, 16..23} (N=lane), lanes 16-31 -> K {8..15, 24..31},
// VGPR v holds K = base + 2(v&3) + 16(v>>2), +1.
// ---------------------------------------------------------------------------
__global__ void wb_reorder(const float* __restrict__ W, unsigned* __restrict__ wb) {
  int id   = blockIdx.x * 256 + threadIdx.x;   // 65536 total
  int v    = id & 7;
  int lane = (id >> 3) & 31;
  int nt   = (id >> 8) & 7;
  int kt   = id >> 11;
  int n    = nt * 16 + (lane & 15);
  int k    = kt * 32 + ((lane >> 4) << 3) + ((v & 3) << 1) + ((v >> 2) << 4);
  bf16 lo = f2bf(W[(size_t)k * D_OUT_C + n]);
  bf16 hi = f2bf(W[(size_t)(k + 1) * D_OUT_C + n]);
  unsigned short l16, h16;
  __builtin_memcpy(&l16, &lo, 2);
  __builtin_memcpy(&h16, &hi, 2);
  wb[id] = (unsigned)l16 | ((unsigned)h16 << 16);
}

// ---------------------------------------------------------------------------
// h = x @ W  via v_wmma_f32_16x16x32_bf16.
// Block: 256 threads = 8 waves; each wave owns a 16(M) x 128(N) strip.
// K streamed in 32-wide tiles; B ktile (8KB) double-buffered through LDS
// (one s_barrier per K step). A fragment built per-lane with v_perm packs.
// A 16x32 bf16 layout: lanes 0-15 hold K {0..7, 16..23}, lanes 16-31 hold
// K {8..15, 24..31} (2 bf16 per VGPR) -> two contiguous float4 pairs per lane.
// ---------------------------------------------------------------------------
__global__ __launch_bounds__(256)
void gemm_h(const float* __restrict__ x, const unsigned* __restrict__ wb,
            float* __restrict__ h, int n) {
  __shared__ unsigned lds_b[2][2048];   // double-buffered ktile: 2 x 8 KB
  const int tid  = threadIdx.x;
  const int lane = tid & 31;
  const int wave = tid >> 5;
  const int m0   = blockIdx.x * 128 + wave * 16;

  int row = m0 + (lane & 15);
  if (row >= n) row = n - 1;                 // clamp (stores are guarded)
  const float* xrow = x + (size_t)row * D_IN_C;
  const int khalf = (lane >> 4) * 8;         // 0 or 8

  v8f acc[8] = {};

  // preload ktile 0
  {
    const uint4* s = (const uint4*)(wb + (size_t)tid * 8);
    uint4* d = (uint4*)&lds_b[0][tid * 8];
    d[0] = s[0];
    d[1] = s[1];
  }
  __syncthreads();

  for (int kt = 0; kt < 32; ++kt) {
    const int cur = kt & 1;
    // stage next B ktile into the other buffer (no readers there this iter)
    if (kt + 1 < 32) {
      const uint4* s = (const uint4*)(wb + (size_t)(kt + 1) * 2048 + tid * 8);
      uint4* d = (uint4*)&lds_b[cur ^ 1][tid * 8];
      d[0] = s[0];
      d[1] = s[1];
    }

    // A fragment (16 bf16 per lane) from two contiguous float4 pairs
    const int k0 = kt * 32 + khalf;
    float4 x0 = *(const float4*)(xrow + k0);
    float4 x1 = *(const float4*)(xrow + k0 + 4);
    float4 x2 = *(const float4*)(xrow + k0 + 16);
    float4 x3 = *(const float4*)(xrow + k0 + 20);
    if (kt + 2 < 32) {
      __builtin_prefetch(xrow + k0 + 64, 0, 0);   // global_prefetch_b8
      __builtin_prefetch(xrow + k0 + 80, 0, 0);
    }
    v8u au;
    au[0] = pack_bf2(x0.x, x0.y);
    au[1] = pack_bf2(x0.z, x0.w);
    au[2] = pack_bf2(x1.x, x1.y);
    au[3] = pack_bf2(x1.z, x1.w);
    au[4] = pack_bf2(x2.x, x2.y);
    au[5] = pack_bf2(x2.z, x2.w);
    au[6] = pack_bf2(x3.x, x3.y);
    au[7] = pack_bf2(x3.z, x3.w);
    v16bf a = __builtin_bit_cast(v16bf, au);

    #pragma unroll
    for (int nt = 0; nt < 8; ++nt) {
      v16bf b = *(const v16bf*)&lds_b[cur][(nt * 32 + lane) * 8];
      acc[nt] = __builtin_amdgcn_wmma_f32_16x16x32_bf16(
          false, a, false, b, (short)0, acc[nt], false, false);
    }
    __syncthreads();   // readers of [cur] and writers of [cur^1] both done
  }

  // C/D layout: VGPR r, lanes 0-15 -> (M=r, N=lane); lanes 16-31 -> (M=8+r)
  const int col0  = lane & 15;
  const int rbase = m0 + ((lane >> 4) << 3);
  if (m0 + 16 <= n) {
    // interior wave: straight-line stores, no exec juggling
    #pragma unroll
    for (int nt = 0; nt < 8; ++nt) {
      #pragma unroll
      for (int r = 0; r < 8; ++r) {
        h[(size_t)(rbase + r) * D_OUT_C + nt * 16 + col0] = acc[nt][r];
      }
    }
  } else {
    #pragma unroll
    for (int nt = 0; nt < 8; ++nt) {
      #pragma unroll
      for (int r = 0; r < 8; ++r) {
        int rr = rbase + r;
        if (rr < n) h[(size_t)rr * D_OUT_C + nt * 16 + col0] = acc[nt][r];
      }
    }
  }
}

// ---------------------------------------------------------------------------
// a_s = h @ att_src, a_d = h @ att_dst  — one wave per node, shfl reduction
// ---------------------------------------------------------------------------
__global__ __launch_bounds__(256)
void att_scores(const float* __restrict__ h, const float* __restrict__ att_s,
                const float* __restrict__ att_d, float* __restrict__ a_s,
                float* __restrict__ a_d, int n) {
  int wave = threadIdx.x >> 5, lane = threadIdx.x & 31;
  int node = blockIdx.x * 8 + wave;
  if (node >= n) return;
  float4 hv = *(const float4*)(h + (size_t)node * D_OUT_C + lane * 4);
  float4 sv = *(const float4*)(att_s + lane * 4);
  float4 dv = *(const float4*)(att_d + lane * 4);
  float ss = hv.x*sv.x + hv.y*sv.y + hv.z*sv.z + hv.w*sv.w;
  float dd = hv.x*dv.x + hv.y*dv.y + hv.z*dv.z + hv.w*dv.w;
  #pragma unroll
  for (int off = 16; off; off >>= 1) {
    ss += __shfl_xor(ss, off, 32);
    dd += __shfl_xor(dd, off, 32);
  }
  if (lane == 0) { a_s[node] = ss; a_d[node] = dd; }
}

// ---------------------------------------------------------------------------
// Edge phase
// ---------------------------------------------------------------------------
__global__ void edge_score(const int* __restrict__ src, const int* __restrict__ dst,
                           const float* __restrict__ a_s, const float* __restrict__ a_d,
                           float* __restrict__ esc, float* __restrict__ smax, int E) {
  int e = blockIdx.x * 256 + threadIdx.x;
  if (e >= E) return;
  float s = a_s[src[e]] + a_d[dst[e]];
  s = s > 0.f ? s : 0.2f * s;            // LeakyReLU(0.2)
  esc[e] = s;
  atomicMaxF(&smax[dst[e]], s);
}

__global__ void edge_exp(const int* __restrict__ dst, float* __restrict__ esc,
                         const float* __restrict__ smax, float* __restrict__ denom, int E) {
  int e = blockIdx.x * 256 + threadIdx.x;
  if (e >= E) return;
  int d = dst[e];
  float ex = __expf(esc[e] - smax[d]);
  esc[e] = ex;
  atomicAddF(&denom[d], ex);
}

// wave per edge: coalesced 128-ch gather + f32 atomic scatter
__global__ __launch_bounds__(256)
void edge_aggregate(const int* __restrict__ src, const int* __restrict__ dst,
                    const float* __restrict__ esc, const float* __restrict__ denom,
                    const float* __restrict__ h, float* __restrict__ outn, int E) {
  int wave = threadIdx.x >> 5, lane = threadIdx.x & 31;
  int e = blockIdx.x * 8 + wave;
  if (e >= E) return;
  int s = src[e], d = dst[e];
  float dn = denom[d];
  dn = dn > 1e-16f ? dn : 1e-16f;
  float alpha = esc[e] / dn;
  float4 hv = *(const float4*)(h + (size_t)s * D_OUT_C + lane * 4);
  float* o = outn + (size_t)d * D_OUT_C + lane * 4;
  atomicAddF(o + 0, alpha * hv.x);
  atomicAddF(o + 1, alpha * hv.y);
  atomicAddF(o + 2, alpha * hv.z);
  atomicAddF(o + 3, alpha * hv.w);
}

// ---------------------------------------------------------------------------
// bias + global mean pool (accumulate)
// ---------------------------------------------------------------------------
__global__ __launch_bounds__(128)
void bias_pool(const float* __restrict__ outn, const float* __restrict__ conv_b,
               const int* __restrict__ batch, float* __restrict__ pooled,
               float* __restrict__ cnt, int n, int colOff, int pIdx) {
  int node = blockIdx.x;
  int ch = threadIdx.x;
  if (node >= n) return;
  float v = outn[(size_t)node * D_OUT_C + ch] + conv_b[ch];
  int g = batch[node];
  atomicAddF(&pooled[(size_t)g * 256 + colOff + ch], v);
  if (ch == 0) atomicAddF(&cnt[pIdx * NGRAPH + g], 1.f);
}

__global__ void pool_finalize(const float* __restrict__ pooled,
                              const float* __restrict__ cnt, float* __restrict__ xc) {
  int i = blockIdx.x * 256 + threadIdx.x;   // 128*256
  if (i >= NGRAPH * 256) return;
  int g = i >> 8, c = i & 255;
  float ct = cnt[(c >> 7) * NGRAPH + g];
  ct = ct > 1.f ? ct : 1.f;
  float v = pooled[i] / ct;
  xc[i] = v > 0.f ? v : 0.01f * v;          // LeakyReLU(0.01)
}

// ---------------------------------------------------------------------------
// MLP: [128,256] -> 256 (lrelu) -> 1 (sigmoid); one block per graph
// ---------------------------------------------------------------------------
__global__ __launch_bounds__(256)
void mlp_head(const float* __restrict__ xc, const float* __restrict__ fc1_w,
              const float* __restrict__ fc1_b, const float* __restrict__ out_w,
              const float* __restrict__ out_b, float* __restrict__ out) {
  __shared__ float hbuf[256];
  int g = blockIdx.x, j = threadIdx.x;
  const float* xr = xc + (size_t)g * 256;
  float acc = fc1_b[j];
  #pragma unroll 4
  for (int k = 0; k < 256; ++k) acc += xr[k] * fc1_w[(size_t)k * 256 + j];
  acc = acc > 0.f ? acc : 0.01f * acc;
  hbuf[j] = acc * out_w[j];
  __syncthreads();
  for (int s2 = 128; s2; s2 >>= 1) {
    if (j < s2) hbuf[j] += hbuf[j + s2];
    __syncthreads();
  }
  if (j == 0) {
    float z = hbuf[0] + out_b[0];
    out[g] = 1.f / (1.f + __expf(-z));
  }
}

// ---------------------------------------------------------------------------
// Launch
// ---------------------------------------------------------------------------
extern "C" void kernel_launch(void* const* d_in, const int* in_sizes, int n_in,
                              void* d_out, int out_size, void* d_ws, size_t ws_size,
                              hipStream_t stream) {
  const int n1 = in_sizes[0] / D_IN_C;
  const int n2 = in_sizes[3] / D_IN_C;
  const int E1 = in_sizes[1] / 2;
  const int E2 = in_sizes[4] / 2;
  const int nMax = n1 > n2 ? n1 : n2;
  const int EMax = E1 > E2 ? E1 : E2;

  const float* W      = (const float*)d_in[6];
  const float* att_s  = (const float*)d_in[7];
  const float* att_d  = (const float*)d_in[8];
  const float* conv_b = (const float*)d_in[9];
  const float* fc1_w  = (const float*)d_in[10];
  const float* fc1_b  = (const float*)d_in[11];
  const float* out_w  = (const float*)d_in[12];
  const float* out_b  = (const float*)d_in[13];

  // workspace carve-up (256B aligned)
  char* ws = (char*)d_ws;
  size_t off = 0;
  auto carve = [&](size_t bytes) {
    char* p = ws + off;
    off += (bytes + 255) & ~size_t(255);
    return p;
  };
  unsigned* wb   = (unsigned*)carve(65536 * sizeof(unsigned));       // 256 KB
  float* h       = (float*)carve((size_t)nMax * D_OUT_C * 4);        // 25.6 MB
  float* outn    = (float*)carve((size_t)nMax * D_OUT_C * 4);        // 25.6 MB
  float* a_s     = (float*)carve((size_t)nMax * 4);
  float* a_d     = (float*)carve((size_t)nMax * 4);
  float* smax    = (float*)carve((size_t)nMax * 4);
  float* denom   = (float*)carve((size_t)nMax * 4);
  float* esc     = (float*)carve((size_t)EMax * 4);                  // 6.4 MB
  float* pooled  = (float*)carve((size_t)NGRAPH * 256 * 4);
  float* cnt     = (float*)carve(2 * NGRAPH * 4);
  float* xc      = (float*)carve((size_t)NGRAPH * 256 * 4);

  // one-time prep
  wb_reorder<<<256, 256, 0, stream>>>(W, wb);
  fill_f<<<(NGRAPH * 256 + 255) / 256, 256, 0, stream>>>(pooled, 0.f, NGRAPH * 256);
  fill_f<<<1, 256, 0, stream>>>(cnt, 0.f, 2 * NGRAPH);

  for (int p = 0; p < 2; ++p) {
    const float* x     = (const float*)d_in[p ? 3 : 0];
    const int*   ei    = (const int*)  d_in[p ? 4 : 1];
    const int*   batch = (const int*)  d_in[p ? 5 : 2];
    const int n = p ? n2 : n1;
    const int E = p ? E2 : E1;
    const int* src = ei;
    const int* dst = ei + E;

    gemm_h<<<(n + 127) / 128, 256, 0, stream>>>(x, wb, h, n);
    att_scores<<<(n + 7) / 8, 256, 0, stream>>>(h, att_s, att_d, a_s, a_d, n);

    fill_f<<<(n + 255) / 256, 256, 0, stream>>>(smax, -INFINITY, n);
    fill_f<<<(n + 255) / 256, 256, 0, stream>>>(denom, 0.f, n);
    fill_f<<<((size_t)n * D_OUT_C + 255) / 256, 256, 0, stream>>>(outn, 0.f, n * D_OUT_C);

    edge_score<<<(E + 255) / 256, 256, 0, stream>>>(src, dst, a_s, a_d, esc, smax, E);
    edge_exp<<<(E + 255) / 256, 256, 0, stream>>>(dst, esc, smax, denom, E);
    edge_aggregate<<<(E + 7) / 8, 256, 0, stream>>>(src, dst, esc, denom, h, outn, E);

    bias_pool<<<n, 128, 0, stream>>>(outn, conv_b, batch, pooled, cnt, n,
                                     p * D_OUT_C, p);
  }

  pool_finalize<<<(NGRAPH * 256 + 255) / 256, 256, 0, stream>>>(pooled, cnt, xc);
  mlp_head<<<NGRAPH, 256, 0, stream>>>(xc, fc1_w, fc1_b, out_w, out_b, (float*)d_out);
}
